// Native_Attention_msa_4664334483726
// MI455X (gfx1250) — matrix operand order repaired and checked by reading, not compiled, not run on hardware
//
#include <hip/hip_runtime.h>
#include <hip/hip_bf16.h>
#include <math.h>

// ---------------------------------------------------------------------------
// Problem constants (B=2, N=2048, C=1024, H=16, hd=64)
// ---------------------------------------------------------------------------
#define BATCH   2
#define SEQ     2048
#define CH      1024
#define HEADS   16
#define HDIM    64
#define BN      (BATCH * SEQ)        // 4096 rows
#define SCALE   0.125f               // hd^-0.5

typedef __attribute__((ext_vector_type(8)))  __bf16 bf16x8;
typedef __attribute__((ext_vector_type(16))) __bf16 bf16x16;
typedef __attribute__((ext_vector_type(8)))  float  floatx8;

// float -> bf16 (round to nearest even), raw bits
__device__ __forceinline__ unsigned short f2bf(float f) {
    unsigned int u = __builtin_bit_cast(unsigned int, f);
    u += 0x7FFFu + ((u >> 16) & 1u);
    return (unsigned short)(u >> 16);
}

// ---------------------------------------------------------------------------
// WMMA fragment loaders (bf16, 16x16x32, layouts per CDNA5 ISA 7.12.2)
// A (16xK=32, row striped within lane):
//   lane<16 : M=lane,    halves = K[k0..k0+7]    and K[k0+16..k0+23]
//   lane>=16: M=lane-16, halves = K[k0+8..k0+15] and K[k0+24..k0+31]
// ---------------------------------------------------------------------------
__device__ __forceinline__ bf16x16
load_a16(const unsigned short* __restrict__ src, int ld, int m0, int k0, int lane) {
    int hf = lane >> 4;
    const unsigned short* p = src + (m0 + (lane & 15)) * ld + k0 + hf * 8;
    bf16x8 lo = *reinterpret_cast<const bf16x8*>(p);
    bf16x8 hi = *reinterpret_cast<const bf16x8*>(p + 16);
    return __builtin_shufflevector(lo, hi, 0,1,2,3,4,5,6,7,8,9,10,11,12,13,14,15);
}

// B (K=32 x 16, column per lane, same K halves as A). Source is row-major with the
// WMMA column index as the row: src[col][k].
__device__ __forceinline__ bf16x16
load_b16(const unsigned short* __restrict__ src, int ld, int n0, int k0, int lane) {
    int hf = lane >> 4;
    const unsigned short* p = src + (n0 + (lane & 15)) * ld + k0 + hf * 8;
    bf16x8 lo = *reinterpret_cast<const bf16x8*>(p);
    bf16x8 hi = *reinterpret_cast<const bf16x8*>(p + 16);
    return __builtin_shufflevector(lo, hi, 0,1,2,3,4,5,6,7,8,9,10,11,12,13,14,15);
}

__device__ __forceinline__ floatx8 wmma_bf16(bf16x16 a, bf16x16 b, floatx8 c) {
    return __builtin_amdgcn_wmma_f32_16x16x32_bf16(false, a, false, b, (short)0, c,
                                                   false, false);
}

// reductions across the 16 lanes holding one C/D row (xor masks stay in-half)
__device__ __forceinline__ float rowmax16(float v) {
    v = fmaxf(v, __shfl_xor(v, 1, 32));
    v = fmaxf(v, __shfl_xor(v, 2, 32));
    v = fmaxf(v, __shfl_xor(v, 4, 32));
    v = fmaxf(v, __shfl_xor(v, 8, 32));
    return v;
}
__device__ __forceinline__ float rowsum16(float v) {
    v += __shfl_xor(v, 1, 32);
    v += __shfl_xor(v, 2, 32);
    v += __shfl_xor(v, 4, 32);
    v += __shfl_xor(v, 8, 32);
    return v;
}

// ---------------------------------------------------------------------------
// Kernel: fp32 -> bf16 conversion (grid-stride)
// ---------------------------------------------------------------------------
__global__ void cvt_bf16_kernel(const float* __restrict__ src,
                                unsigned short* __restrict__ dst, int n) {
    int i = blockIdx.x * blockDim.x + threadIdx.x;
    int stride = gridDim.x * blockDim.x;
    for (; i < n; i += stride) dst[i] = f2bf(src[i]);
}

// ---------------------------------------------------------------------------
// Kernel: QKV GEMM (out = X @ Wqkv^T), 128x64 tiles (wave = 32 rows), K=1024,
// software-pipelined K loop (double-buffered fragments).
// Epilogue: L2-normalize q/k rows over hd=64, emit bf16:
//   q,k : [B,H,N,64] row-major;  v : [B,H,64,N] (transposed for PV WMMA)
// ---------------------------------------------------------------------------
__global__ void __launch_bounds__(128)
qkv_gemm_norm_kernel(const unsigned short* __restrict__ xb,
                     const unsigned short* __restrict__ wb,
                     unsigned short* __restrict__ qbuf,
                     unsigned short* __restrict__ kbuf,
                     unsigned short* __restrict__ vtbuf) {
    __shared__ float tile[128][65];
    __shared__ float invn[128];

    int bid    = blockIdx.x;                 // 32 * 48 tiles
    int tile_m = (bid / 48) * 128;
    int tile_d = (bid % 48) * 64;
    int tid    = threadIdx.x;
    int w      = tid >> 5;
    int lane   = tid & 31;
    int hf     = lane >> 4;
    int m0     = tile_m + w * 32;

    floatx8 acc[2][4] = {};

    // pipelined K loop: cur fragments in (a0,a1,b0..b3); next loaded ahead
    bf16x16 a0 = load_a16(xb, CH, m0,      0, lane);
    bf16x16 a1 = load_a16(xb, CH, m0 + 16, 0, lane);
    bf16x16 b0 = load_b16(wb, CH, tile_d,      0, lane);
    bf16x16 b1 = load_b16(wb, CH, tile_d + 16, 0, lane);
    bf16x16 b2 = load_b16(wb, CH, tile_d + 32, 0, lane);
    bf16x16 b3 = load_b16(wb, CH, tile_d + 48, 0, lane);

    auto compute = [&]() {
        acc[0][0] = wmma_bf16(a0, b0, acc[0][0]);
        acc[0][1] = wmma_bf16(a0, b1, acc[0][1]);
        acc[0][2] = wmma_bf16(a0, b2, acc[0][2]);
        acc[0][3] = wmma_bf16(a0, b3, acc[0][3]);
        acc[1][0] = wmma_bf16(a1, b0, acc[1][0]);
        acc[1][1] = wmma_bf16(a1, b1, acc[1][1]);
        acc[1][2] = wmma_bf16(a1, b2, acc[1][2]);
        acc[1][3] = wmma_bf16(a1, b3, acc[1][3]);
    };

    for (int kk = 32; kk < CH; kk += 32) {
        bf16x16 na0 = load_a16(xb, CH, m0,      kk, lane);
        bf16x16 na1 = load_a16(xb, CH, m0 + 16, kk, lane);
        bf16x16 nb0 = load_b16(wb, CH, tile_d,      kk, lane);
        bf16x16 nb1 = load_b16(wb, CH, tile_d + 16, kk, lane);
        bf16x16 nb2 = load_b16(wb, CH, tile_d + 32, kk, lane);
        bf16x16 nb3 = load_b16(wb, CH, tile_d + 48, kk, lane);
        compute();
        a0 = na0; a1 = na1; b0 = nb0; b1 = nb1; b2 = nb2; b3 = nb3;
    }
    compute();

#pragma unroll
    for (int i = 0; i < 2; ++i)
#pragma unroll
        for (int j = 0; j < 4; ++j)
#pragma unroll
            for (int r = 0; r < 8; ++r)
                tile[w * 32 + i * 16 + r + hf * 8][16 * j + (lane & 15)] = acc[i][j][r];
    __syncthreads();

    int t = tile_d >> 10;            // 0:q 1:k 2:v
    int h = (tile_d >> 6) & 15;

    if (t < 2) {                      // uniform branch per block
        float s = 0.f;
        for (int c = 0; c < 64; ++c) { float v = tile[tid][c]; s += v * v; }
        invn[tid] = 1.0f / (sqrtf(s) + 1e-8f);
        __syncthreads();

        unsigned short* dst = (t == 0) ? qbuf : kbuf;
        for (int idx = tid; idx < 128 * 64; idx += 128) {
            int row = idx >> 6, e = idx & 63;   // lanes -> consecutive e
            int ng = tile_m + row;
            int b = ng >> 11, n = ng & 2047;
            float v = tile[row][e] * invn[row];
            dst[((b * HEADS + h) * SEQ + n) * HDIM + e] = f2bf(v);
        }
    } else {
        for (int idx = tid; idx < 128 * 64; idx += 128) {
            int row = idx & 127, e = idx >> 7;  // lanes -> consecutive n
            int ng = tile_m + row;
            int b = ng >> 11, n = ng & 2047;
            vtbuf[((b * HEADS + h) * HDIM + e) * SEQ + n] = f2bf(tile[row][e]);
        }
    }
}

// ---------------------------------------------------------------------------
// Kernel: attention with online softmax.
// One wave owns 16 query rows; streams 32-key blocks:
//   V frags loaded first (overlap softmax), S = q k^T (4 WMMAs),
//   K frags for the NEXT block refilled right after consumption,
//   logit = mask ? S*scale*s_j : 0, online softmax,
//   P staged through LDS (s_wait_dscnt), O += P V (4 WMMAs).
// ---------------------------------------------------------------------------
__global__ void __launch_bounds__(128)
attn_kernel(const unsigned short* __restrict__ qbuf,
            const unsigned short* __restrict__ kbuf,
            const unsigned short* __restrict__ vtbuf,
            const float* __restrict__ score,
            const int* __restrict__ umptr,
            unsigned short* __restrict__ aout) {
    __shared__ __align__(16) unsigned short pshare[4][512]; // per-wave 16x32 bf16

    int bid   = blockIdx.x;           // B*H*32
    int qtile = bid & 31;
    int bh    = bid >> 5;
    int b     = bh >> 4;
    int h     = bh & 15;
    int tid   = threadIdx.x;
    int w     = tid >> 5;
    int lane  = tid & 31;
    int hf    = lane >> 4;
    int um    = umptr[0];

    const unsigned short* qb = qbuf + bh * SEQ * HDIM;
    const unsigned short* kb = kbuf + bh * SEQ * HDIM;
    const unsigned short* vt = vtbuf + bh * HDIM * SEQ;
    unsigned short* ps = pshare[w];

    int q0 = qtile * 64 + w * 16;
    bf16x16 qf0 = load_a16(qb, HDIM, q0, 0, lane);
    bf16x16 qf1 = load_a16(qb, HDIM, q0, 32, lane);

    floatx8 acc_o[4] = {};
    float m_r[8], l_r[8], si[8];
#pragma unroll
    for (int r = 0; r < 8; ++r) {
        m_r[r] = -3.0e38f; l_r[r] = 0.f;
        si[r] = score[q0 + r + hf * 8];
    }

    // preload K fragments for the first key block
    bf16x16 kf0 = load_b16(kb, HDIM, 0,  0,  lane);
    bf16x16 kf1 = load_b16(kb, HDIM, 0,  32, lane);
    bf16x16 kf2 = load_b16(kb, HDIM, 16, 0,  lane);
    bf16x16 kf3 = load_b16(kb, HDIM, 16, 32, lane);

    for (int j0 = 0; j0 < SEQ; j0 += 32) {
        // V fragments issued first: overlap the softmax VALU section
        bf16x16 bv0 = load_b16(vt, SEQ, 0,  j0, lane);
        bf16x16 bv1 = load_b16(vt, SEQ, 16, j0, lane);
        bf16x16 bv2 = load_b16(vt, SEQ, 32, j0, lane);
        bf16x16 bv3 = load_b16(vt, SEQ, 48, j0, lane);

        floatx8 s0 = {}, s1 = {};
        s0 = wmma_bf16(qf0, kf0, s0);
        s0 = wmma_bf16(qf1, kf1, s0);
        s1 = wmma_bf16(qf0, kf2, s1);
        s1 = wmma_bf16(qf1, kf3, s1);

        // refill K fragments for the next block: in flight across softmax + PV
        if (j0 + 32 < SEQ) {
            kf0 = load_b16(kb, HDIM, j0 + 32, 0,  lane);
            kf1 = load_b16(kb, HDIM, j0 + 32, 32, lane);
            kf2 = load_b16(kb, HDIM, j0 + 48, 0,  lane);
            kf3 = load_b16(kb, HDIM, j0 + 48, 32, lane);
        }

        float sj0 = score[j0 + (lane & 15)];
        float sj1 = score[j0 + 16 + (lane & 15)];

        float p0[8], p1[8];
#pragma unroll
        for (int r = 0; r < 8; ++r) {
            float v0 = s0[r] * SCALE * sj0;
            float v1 = s1[r] * SCALE * sj1;
            if (um) {   // masked logits become 0 (reference multiplies pre-softmax)
                if (!(sj0 > si[r] - 0.1f)) v0 = 0.f;
                if (!(sj1 > si[r] - 0.1f)) v1 = 0.f;
            }
            float rm = rowmax16(fmaxf(v0, v1));
            float mn = fmaxf(m_r[r], rm);
            float al = __expf(m_r[r] - mn);
            float e0 = __expf(v0 - mn);
            float e1 = __expf(v1 - mn);
            float rs = rowsum16(e0 + e1);
            l_r[r] = l_r[r] * al + rs;
            m_r[r] = mn;
            p0[r] = e0; p1[r] = e1;
#pragma unroll
            for (int j = 0; j < 4; ++j) acc_o[j][r] *= al;
        }

        // stage P (C/D layout -> row-major 16x32 bf16 in LDS)
#pragma unroll
        for (int r = 0; r < 8; ++r) {
            int row = r + hf * 8;
            ps[row * 32 + (lane & 15)]      = f2bf(p0[r]);
            ps[row * 32 + 16 + (lane & 15)] = f2bf(p1[r]);
        }
        asm volatile("s_wait_dscnt 0" ::: "memory");  // CDNA5 split DS counter

        // reload P as an A fragment (16x32)
        bf16x16 ap;
        {
            const unsigned short* pp = ps + (lane & 15) * 32 + hf * 8;
            bf16x8 lo = *reinterpret_cast<const bf16x8*>(pp);
            bf16x8 hi = *reinterpret_cast<const bf16x8*>(pp + 16);
            ap = __builtin_shufflevector(lo, hi, 0,1,2,3,4,5,6,7,8,9,10,11,12,13,14,15);
        }

        // O += P @ V   (V transposed: vt[e][j], contiguous in j)
        acc_o[0] = wmma_bf16(ap, bv0, acc_o[0]);
        acc_o[1] = wmma_bf16(ap, bv1, acc_o[1]);
        acc_o[2] = wmma_bf16(ap, bv2, acc_o[2]);
        acc_o[3] = wmma_bf16(ap, bv3, acc_o[3]);
    }

    float inv_l[8];
#pragma unroll
    for (int r = 0; r < 8; ++r) inv_l[r] = 1.0f / l_r[r];

    // write attention output as bf16 [B*N, C], column block = h*64
#pragma unroll
    for (int j = 0; j < 4; ++j)
#pragma unroll
        for (int r = 0; r < 8; ++r) {
            int nrow = b * SEQ + q0 + r + hf * 8;
            int col  = h * HDIM + 16 * j + (lane & 15);
            aout[nrow * CH + col] = f2bf(acc_o[j][r] * inv_l[r]);
        }
}

// ---------------------------------------------------------------------------
// Kernel: projection GEMM (out = A @ Wp^T + bias), 128x64 tiles, pipelined,
// fp32 output
// ---------------------------------------------------------------------------
__global__ void __launch_bounds__(128)
proj_gemm_kernel(const unsigned short* __restrict__ ab,
                 const unsigned short* __restrict__ wb,
                 const float* __restrict__ bias,
                 float* __restrict__ out) {
    int bid    = blockIdx.x;              // 32 * 16 tiles
    int tile_m = (bid >> 4) * 128;
    int tile_d = (bid & 15) * 64;
    int tid    = threadIdx.x;
    int w      = tid >> 5;
    int lane   = tid & 31;
    int hf     = lane >> 4;
    int m0     = tile_m + w * 32;

    floatx8 acc[2][4] = {};

    bf16x16 a0 = load_a16(ab, CH, m0,      0, lane);
    bf16x16 a1 = load_a16(ab, CH, m0 + 16, 0, lane);
    bf16x16 b0 = load_b16(wb, CH, tile_d,      0, lane);
    bf16x16 b1 = load_b16(wb, CH, tile_d + 16, 0, lane);
    bf16x16 b2 = load_b16(wb, CH, tile_d + 32, 0, lane);
    bf16x16 b3 = load_b16(wb, CH, tile_d + 48, 0, lane);

    auto compute = [&]() {
        acc[0][0] = wmma_bf16(a0, b0, acc[0][0]);
        acc[0][1] = wmma_bf16(a0, b1, acc[0][1]);
        acc[0][2] = wmma_bf16(a0, b2, acc[0][2]);
        acc[0][3] = wmma_bf16(a0, b3, acc[0][3]);
        acc[1][0] = wmma_bf16(a1, b0, acc[1][0]);
        acc[1][1] = wmma_bf16(a1, b1, acc[1][1]);
        acc[1][2] = wmma_bf16(a1, b2, acc[1][2]);
        acc[1][3] = wmma_bf16(a1, b3, acc[1][3]);
    };

    for (int kk = 32; kk < CH; kk += 32) {
        bf16x16 na0 = load_a16(ab, CH, m0,      kk, lane);
        bf16x16 na1 = load_a16(ab, CH, m0 + 16, kk, lane);
        bf16x16 nb0 = load_b16(wb, CH, tile_d,      kk, lane);
        bf16x16 nb1 = load_b16(wb, CH, tile_d + 16, kk, lane);
        bf16x16 nb2 = load_b16(wb, CH, tile_d + 32, kk, lane);
        bf16x16 nb3 = load_b16(wb, CH, tile_d + 48, kk, lane);
        compute();
        a0 = na0; a1 = na1; b0 = nb0; b1 = nb1; b2 = nb2; b3 = nb3;
    }
    compute();

#pragma unroll
    for (int i = 0; i < 2; ++i)
#pragma unroll
        for (int j = 0; j < 4; ++j) {
            int col = tile_d + 16 * j + (lane & 15);
            float bv = bias[col];
#pragma unroll
            for (int r = 0; r < 8; ++r)
                out[(m0 + i * 16 + r + hf * 8) * CH + col] = acc[i][j][r] + bv;
        }
}

// ---------------------------------------------------------------------------
// Host launcher
// ---------------------------------------------------------------------------
extern "C" void kernel_launch(void* const* d_in, const int* in_sizes, int n_in,
                              void* d_out, int out_size, void* d_ws, size_t ws_size,
                              hipStream_t stream) {
    const float* x_cls     = (const float*)d_in[0];   // [B,N,C]
    const float* cls_score = (const float*)d_in[1];   // [N]
    const float* qkv_w     = (const float*)d_in[2];   // [3C,C]
    const float* proj_w    = (const float*)d_in[3];   // [C,C]
    const float* proj_b    = (const float*)d_in[4];   // [C]
    const int*   use_mask  = (const int*)d_in[5];     // scalar
    float*       out       = (float*)d_out;           // [B,N,C]

    char* ws = (char*)d_ws;
    unsigned short* x_bf    = (unsigned short*)(ws);                     //  8 MB
    unsigned short* qkvw_bf = (unsigned short*)(ws + (8u  << 20));       //  6 MB
    unsigned short* projw_bf= (unsigned short*)(ws + (14u << 20));       //  2 MB
    unsigned short* qbuf    = (unsigned short*)(ws + (16u << 20));       //  8 MB
    unsigned short* kbuf    = (unsigned short*)(ws + (24u << 20));       //  8 MB
    unsigned short* vtbuf   = (unsigned short*)(ws + (32u << 20));       //  8 MB
    unsigned short* aout    = (unsigned short*)(ws + (40u << 20));       //  8 MB

    // 1) fp32 -> bf16 conversions
    cvt_bf16_kernel<<<2048, 256, 0, stream>>>(x_cls,  x_bf,    BN * CH);
    cvt_bf16_kernel<<<2048, 256, 0, stream>>>(qkv_w,  qkvw_bf, 3 * CH * CH);
    cvt_bf16_kernel<<<1024, 256, 0, stream>>>(proj_w, projw_bf, CH * CH);

    // 2) QKV GEMM + q/k L2 normalization + layout (q,k row-major; v transposed)
    qkv_gemm_norm_kernel<<<(BN / 128) * 48, 128, 0, stream>>>(
        x_bf, qkvw_bf, qbuf, kbuf, vtbuf);

    // 3) cosine attention with masked logits + online softmax
    attn_kernel<<<BATCH * HEADS * (SEQ / 64), 128, 0, stream>>>(
        qbuf, kbuf, vtbuf, cls_score, use_mask, aout);

    // 4) output projection + bias
    proj_gemm_kernel<<<(BN / 128) * (CH / 64), 128, 0, stream>>>(
        aout, projw_bf, proj_b, out);
}
